// Grid_INR2D_27736898797776
// MI455X (gfx1250) — matrix operand order repaired and checked by reading, not compiled, not run on hardware
//
#include <hip/hip_runtime.h>
#include <hip/hip_bf16.h>

// ---------------------------------------------------------------------------
// Grid-INR 2D fused kernel for gfx1250 (MI455X).
//   feat: (8,32,256,256) f32;  g0..g3: (16,G,G) f32, G=32/64/128/256
//   MLP: 96 -> 64 (relu) -> 64 (relu) -> 3, weights f32.
// Strategy: f16 WMMA (v_wmma_f32_16x16x32_f16) for all three layers;
// batch-invariant grid-sample precomputed once into d_ws as f16 pixel-major,
// then staged into LDS with async-to-LDS copies (ASYNCcnt path).
// ---------------------------------------------------------------------------

typedef __attribute__((ext_vector_type(8)))  _Float16 h8_t;
typedef __attribute__((ext_vector_type(16))) _Float16 h16_t;
typedef __attribute__((ext_vector_type(8)))  float    f8_t;

constexpr int HW      = 256 * 256;   // pixels per image
constexpr int C_FEAT  = 32;
constexpr int C_LOC   = 64;
constexpr int K_IN    = 96;          // C_FEAT + C_LOC
constexpr int HID     = 64;
constexpr int XSTR    = 104;         // LDS stride (halves) for X tile: 208B, 16B-aligned
constexpr int HSTR    = 72;          // LDS stride (halves) for hidden: 144B, 16B-aligned
constexpr int TSTR    = 72;          // LDS stride (halves) in precompute transpose

// workspace layout (bytes)
constexpr size_t LOCAL_BYTES = (size_t)HW * C_LOC * 2;          // 8 MiB f16, [pix][64]
constexpr size_t W1_OFF = LOCAL_BYTES;                          // 64x96 f16
constexpr size_t W2_OFF = W1_OFF + (size_t)HID * K_IN * 2;      // 64x64 f16
constexpr size_t W3_OFF = W2_OFF + (size_t)HID * HID * 2;       // 16x64 f16 (zero-padded)

// ---------------------------------------------------------------------------
__global__ __launch_bounds__(256) void prep_weights(
    const float* __restrict__ w1, const float* __restrict__ w2,
    const float* __restrict__ w3,
    _Float16* __restrict__ w1h, _Float16* __restrict__ w2h,
    _Float16* __restrict__ w3h)
{
    for (int i = threadIdx.x; i < HID * K_IN; i += 256) w1h[i] = (_Float16)w1[i];
    for (int i = threadIdx.x; i < HID * HID;  i += 256) w2h[i] = (_Float16)w2[i];
    for (int i = threadIdx.x; i < 16 * HID;   i += 256)
        w3h[i] = (i < 3 * HID) ? (_Float16)w3[i] : (_Float16)0.0f;
}

// ---------------------------------------------------------------------------
// Bilinear sample of one pyramid level at one pixel center: the corner
// indices/weights are shared by all 16 channels of the level.
__device__ __forceinline__ void sample_level(const float* __restrict__ g, int G,
                                             int x, int y, _Float16* dst)
{
    // xs[i] = (2i+1)/256 - 1 ;  ix = (xs+1)*0.5*(G-1) = (2i+1)*(G-1)/512
    float scale = (float)(G - 1) * (1.0f / 512.0f);
    float ix = (2.0f * (float)x + 1.0f) * scale;
    float iy = (2.0f * (float)y + 1.0f) * scale;
    int ix0 = (int)floorf(ix); int iy0 = (int)floorf(iy);
    ix0 = min(max(ix0, 0), G - 1); iy0 = min(max(iy0, 0), G - 1);
    int ix1 = min(ix0 + 1, G - 1); int iy1 = min(iy0 + 1, G - 1);
    float wx = ix - (float)ix0, wy = iy - (float)iy0;
    float w00 = (1.f - wy) * (1.f - wx), w01 = (1.f - wy) * wx;
    float w10 = wy * (1.f - wx),         w11 = wy * wx;
    int i00 = iy0 * G + ix0, i01 = iy0 * G + ix1;
    int i10 = iy1 * G + ix0, i11 = iy1 * G + ix1;
    int GG = G * G;
    for (int cc = 0; cc < 16; ++cc) {
        const float* p = g + cc * GG;
        float r = p[i00] * w00 + p[i01] * w01 + p[i10] * w10 + p[i11] * w11;
        dst[cc] = (_Float16)r;
    }
}

// One thread per pixel; results transposed through LDS so the pixel-major
// [pix][64] f16 buffer is written with fully coalesced b128 stores.
__global__ __launch_bounds__(256) void precompute_local(
    const float* __restrict__ g0, const float* __restrict__ g1,
    const float* __restrict__ g2, const float* __restrict__ g3,
    _Float16* __restrict__ localh)
{
    __shared__ _Float16 T[256 * TSTR];
    const int tid = threadIdx.x;
    const int pix = blockIdx.x * 256 + tid;
    const int x = pix & 255, y = pix >> 8;

    _Float16* row = &T[tid * TSTR];
    sample_level(g0, 32,  x, y, row + 0);
    sample_level(g1, 64,  x, y, row + 16);
    sample_level(g2, 128, x, y, row + 32);
    sample_level(g3, 256, x, y, row + 48);
    __syncthreads();

    // 256 px * 64 halves = 2048 x 16B chunks; consecutive threads -> consecutive
    // global chunks (coalesced b128 stores).
    _Float16* gbase = localh + (size_t)blockIdx.x * 256 * C_LOC;
#pragma unroll
    for (int q = 0; q < 8; ++q) {
        int ch = q * 256 + tid;             // 0..2047
        int px = ch >> 3, ci = ch & 7;
        h8_t v = *(const h8_t*)&T[px * TSTR + ci * 8];
        *(h8_t*)(gbase + (size_t)ch * 8) = v;
    }
}

// ---------------------------------------------------------------------------
__device__ __forceinline__ h16_t cat8(h8_t lo, h8_t hi) {
    return __builtin_shufflevector(lo, hi, 0,1,2,3,4,5,6,7,8,9,10,11,12,13,14,15);
}

// 16-bit A fragment (16 rows x 32 K), ISA 7.12.2 layout:
// lanes 0-15: m=lane, K = kb+[0..7] and kb+[16..23]
// lanes 16-31: m=lane-16, K = kb+[8..15] and kb+[24..31]
__device__ __forceinline__ h16_t load_A_lds(const _Float16* base, int stride,
                                            int row0, int kb, int lane) {
    const _Float16* p = base + (row0 + (lane & 15)) * stride + kb + ((lane >> 4) << 3);
    h8_t lo = *(const h8_t*)p;
    h8_t hi = *(const h8_t*)(p + 16);
    return cat8(lo, hi);
}

// 16-bit B fragment (32 K x 16 cols): lanes 0-15 hold n=n0+lane with K=kb+[0..7],kb+[16..23];
// lanes 16-31 hold n with K=kb+[8..15],kb+[24..31]. w is [N][kstride] f16 row-major.
__device__ __forceinline__ h16_t load_B_w(const _Float16* __restrict__ w, int kstride,
                                          int n0, int kb, int lane) {
    const _Float16* p = w + (size_t)(n0 + (lane & 15)) * kstride + kb + ((lane >> 4) << 3);
    h8_t lo = *(const h8_t*)p;
    h8_t hi = *(const h8_t*)(p + 16);
    return cat8(lo, hi);
}

#define WMMA_F16(Afrag, Bfrag, Cacc) \
    __builtin_amdgcn_wmma_f32_16x16x32_f16(false, (Afrag), false, (Bfrag), (short)0, (Cacc), false, false)

// ---------------------------------------------------------------------------
// One block = 256 threads = 8 waves; block owns 128 contiguous pixels of one
// batch image; each wave owns 16 pixels (the WMMA M dimension).
__global__ __launch_bounds__(256) void inr_mlp(
    const float*    __restrict__ feat,
    const _Float16* __restrict__ localh,   // [pix][64] f16
    const _Float16* __restrict__ w1h,
    const _Float16* __restrict__ w2h,
    const _Float16* __restrict__ w3h,
    const float*    __restrict__ b1,
    const float*    __restrict__ b2,
    const float*    __restrict__ b3,
    float*          __restrict__ out)
{
    __shared__ _Float16 Xl[128 * XSTR];   // [pixel][96] inputs, f16
    __shared__ _Float16 Hl[128 * HSTR];   // [pixel][64] hidden, f16 (per-wave rows)

    const int tid = threadIdx.x;
    const int b   = blockIdx.x >> 9;            // 512 blocks per batch image
    const int p0  = (blockIdx.x & 511) * 128;   // pixel base of this tile

    // ---- stage grid-sample channels with async global->LDS b128 copies ----
    // Tile's local block: 128 px x 128B, row px lands at Xl[px][32..95].
    {
        unsigned ldsbase = (unsigned)(uintptr_t)(&Xl[0]);   // LDS byte offset
        unsigned long long gbase =
            (unsigned long long)(const char*)localh + (unsigned long long)p0 * (C_LOC * 2);
#pragma unroll
        for (int q = 0; q < 4; ++q) {
            int ch = q * 256 + tid;             // 0..1023 16B chunks
            int px = ch >> 3, ci = ch & 7;
            unsigned lds_off = ldsbase + (unsigned)(px * XSTR + C_FEAT) * 2u + (unsigned)ci * 16u;
            unsigned long long gsrc = gbase + (unsigned long long)px * (C_LOC * 2)
                                            + (unsigned long long)ci * 16u;
            asm volatile("global_load_async_to_lds_b128 %0, %1, off"
                         :: "v"(lds_off), "v"(gsrc) : "memory");
        }
    }

    // ---- stage feat channels (f32 -> f16), 32c x 128px, overlaps async ----
    {
        const int c = tid >> 3, sub = tid & 7;  // 8 threads per channel
        const float* src = feat + ((size_t)b * C_FEAT + c) * HW + p0 + sub * 16;
#pragma unroll
        for (int q = 0; q < 4; ++q) {
            float4 v = *(const float4*)(src + q * 4);
            int px = sub * 16 + q * 4;
            Xl[(px + 0) * XSTR + c] = (_Float16)v.x;
            Xl[(px + 1) * XSTR + c] = (_Float16)v.y;
            Xl[(px + 2) * XSTR + c] = (_Float16)v.z;
            Xl[(px + 3) * XSTR + c] = (_Float16)v.w;
        }
    }
    asm volatile("s_wait_asynccnt 0x0" ::: "memory");
    __syncthreads();

    const int lane = tid & 31;          // wave32
    const int px0  = (tid >> 5) * 16;   // this wave's 16-pixel row block

    // ---- layer 1: 96 -> 64, 3 k-steps x 4 n-tiles ----
    h16_t a0 = load_A_lds(Xl, XSTR, px0, 0,  lane);
    h16_t a1 = load_A_lds(Xl, XSTR, px0, 32, lane);
    h16_t a2 = load_A_lds(Xl, XSTR, px0, 64, lane);
#pragma unroll
    for (int nt = 0; nt < 4; ++nt) {
        f8_t acc = {};
        acc = WMMA_F16(a0, load_B_w(w1h, K_IN, nt * 16, 0,  lane), acc);
        acc = WMMA_F16(a1, load_B_w(w1h, K_IN, nt * 16, 32, lane), acc);
        acc = WMMA_F16(a2, load_B_w(w1h, K_IN, nt * 16, 64, lane), acc);
        float bias = b1[nt * 16 + (lane & 15)];
#pragma unroll
        for (int v = 0; v < 8; ++v) {
            // D layout: VGPR v -> M = v + 8*(lane>=16), N = lane&15
            float val = fmaxf(acc[v] + bias, 0.0f);
            Hl[(px0 + v + ((lane >> 4) << 3)) * HSTR + nt * 16 + (lane & 15)] = (_Float16)val;
        }
    }

    // ---- layer 2: 64 -> 64, 2 k-steps x 4 n-tiles (in-place via regs) ----
    h16_t c0 = load_A_lds(Hl, HSTR, px0, 0,  lane);
    h16_t c1 = load_A_lds(Hl, HSTR, px0, 32, lane);
#pragma unroll
    for (int nt = 0; nt < 4; ++nt) {
        f8_t acc = {};
        acc = WMMA_F16(c0, load_B_w(w2h, HID, nt * 16, 0,  lane), acc);
        acc = WMMA_F16(c1, load_B_w(w2h, HID, nt * 16, 32, lane), acc);
        float bias = b2[nt * 16 + (lane & 15)];
#pragma unroll
        for (int v = 0; v < 8; ++v) {
            float val = fmaxf(acc[v] + bias, 0.0f);
            Hl[(px0 + v + ((lane >> 4) << 3)) * HSTR + nt * 16 + (lane & 15)] = (_Float16)val;
        }
    }

    // ---- layer 3: 64 -> 3 (padded to 16), 2 k-steps x 1 n-tile ----
    h16_t d0 = load_A_lds(Hl, HSTR, px0, 0,  lane);
    h16_t d1 = load_A_lds(Hl, HSTR, px0, 32, lane);
    f8_t acc = {};
    acc = WMMA_F16(d0, load_B_w(w3h, HID, 0, 0,  lane), acc);
    acc = WMMA_F16(d1, load_B_w(w3h, HID, 0, 32, lane), acc);

    int n = lane & 15;
    if (n < 3) {
        float bias = b3[n];
        float* op = out + ((size_t)b * 3 + n) * HW + p0 + px0 + ((lane >> 4) << 3);
#pragma unroll
        for (int v = 0; v < 8; ++v)   // 8 consecutive pixels per lane -> 32B stores
            op[v] = acc[v] + bias;
    }
}

// ---------------------------------------------------------------------------
extern "C" void kernel_launch(void* const* d_in, const int* in_sizes, int n_in,
                              void* d_out, int out_size, void* d_ws, size_t ws_size,
                              hipStream_t stream) {
    const float* feat = (const float*)d_in[0];
    const float* g0   = (const float*)d_in[1];
    const float* g1   = (const float*)d_in[2];
    const float* g2   = (const float*)d_in[3];
    const float* g3   = (const float*)d_in[4];
    const float* w1   = (const float*)d_in[5];
    const float* b1   = (const float*)d_in[6];
    const float* w2   = (const float*)d_in[7];
    const float* b2   = (const float*)d_in[8];
    const float* w3   = (const float*)d_in[9];
    const float* b3   = (const float*)d_in[10];
    float* out = (float*)d_out;

    char* ws = (char*)d_ws;                         // needs ~8.5 MB
    _Float16* localh = (_Float16*)ws;
    _Float16* w1h    = (_Float16*)(ws + W1_OFF);
    _Float16* w2h    = (_Float16*)(ws + W2_OFF);
    _Float16* w3h    = (_Float16*)(ws + W3_OFF);

    prep_weights<<<1, 256, 0, stream>>>(w1, w2, w3, w1h, w2h, w3h);
    precompute_local<<<HW / 256, 256, 0, stream>>>(g0, g1, g2, g3, localh);
    inr_mlp<<<8 * (HW / 128), 256, 0, stream>>>(feat, localh, w1h, w2h, w3h,
                                                b1, b2, b3, out);
}
